// HeteroGAT_1795296329874
// MI455X (gfx1250) — matrix operand order repaired and checked by reading, not compiled, not run on hardware
//
#include <hip/hip_runtime.h>

typedef unsigned short u16;
typedef __attribute__((ext_vector_type(16))) __bf16 v16bf;
typedef __attribute__((ext_vector_type(8)))  float  v8f;

#define NEG_SLOPE 0.2f

// ---------------------------------------------------------------- utilities

__device__ __forceinline__ u16 f2bf_bits(float f) {
    unsigned u = __float_as_uint(f);
    unsigned r = (u + 0x7FFFu + ((u >> 16) & 1u)) >> 16;   // round-to-nearest-even
    return (u16)r;
}

__device__ __forceinline__ float atomicMaxF32(float* addr, float val) {
    // sign-aware int/uint atomic trick -> native global_atomic_max_i32 / min_u32
    if (val >= 0.0f)
        return __int_as_float(atomicMax((int*)addr, __float_as_int(val)));
    else
        return __uint_as_float(atomicMin((unsigned int*)addr, __float_as_uint(val)));
}

// ---------------------------------------------------------------- kernels

__global__ void k_cast_bf16(const float* __restrict__ in, u16* __restrict__ out, long n) {
    long i = (long)blockIdx.x * blockDim.x + threadIdx.x;
    if (i < n) out[i] = f2bf_bits(in[i]);
}

// W [K,F] f32 -> WT [F,K] bf16 (transpose+cast so B-fragments load contiguously)
__global__ void k_cast_transpose_bf16(const float* __restrict__ W, u16* __restrict__ WT,
                                      int K, int F) {
    long i = (long)blockIdx.x * blockDim.x + threadIdx.x;
    if (i >= (long)K * F) return;
    int k = (int)(i / F);
    int f = (int)(i - (long)k * F);
    WT[(size_t)f * K + k] = f2bf_bits(W[i]);
}

__global__ void k_fill_f32(float* __restrict__ p, float v, long n) {
    long i = (long)blockIdx.x * blockDim.x + threadIdx.x;
    if (i < n) p[i] = v;
}

// C[M,Nc] (f32) = A[M,K] (bf16 row-major) x B (given as BT[Nc,K] bf16 row-major) + bias
// one wave32 computes a 16x64 tile: A-frag loaded once per k-step, 4 WMMAs.
__global__ void k_gemm_bf16_wmma(const u16* __restrict__ A, const u16* __restrict__ BT,
                                 const float* __restrict__ bias, float* __restrict__ C,
                                 int M, int K, int Nc) {
    constexpr int NT = 4;                       // 4 x 16 = 64 output columns per wave
    int lane   = threadIdx.x & 31;
    int wave   = blockIdx.x * (blockDim.x >> 5) + (threadIdx.x >> 5);
    int tilesN = Nc >> 6;
    int tilesM = M >> 4;
    if (wave >= tilesM * tilesN) return;
    int tm = wave / tilesN;
    int tn = wave - tm * tilesN;

    int r    = lane & 15;   // M index (A) / N index (B,C)
    int half = lane >> 4;

    const u16* arow = A + (size_t)(tm * 16 + r) * K;
    const u16* brow[NT];
#pragma unroll
    for (int t = 0; t < NT; ++t)
        brow[t] = BT + (size_t)(tn * 64 + t * 16 + r) * K;

    v8f acc[NT];
#pragma unroll
    for (int t = 0; t < NT; ++t)
        acc[t] = (v8f){0.f, 0.f, 0.f, 0.f, 0.f, 0.f, 0.f, 0.f};

    for (int k0 = 0; k0 < K; k0 += 32) {
        // A fragment (ISA 16-bit A 16x32 table): VGPR v packs K = (v%4)*2 + (v/4)*16 + half*8
        v16bf av;
#pragma unroll
        for (int v = 0; v < 8; ++v) {
            int kk = k0 + ((v & 3) << 1) + ((v >> 2) << 4) + (half << 3);
            unsigned aw = *(const unsigned*)(arow + kk);
            av[2 * v]     = __builtin_bit_cast(__bf16, (u16)(aw & 0xFFFFu));
            av[2 * v + 1] = __builtin_bit_cast(__bf16, (u16)(aw >> 16));
        }
#pragma unroll
        for (int t = 0; t < NT; ++t) {
            // B fragment (same K order as A): VGPR v packs K = (v%4)*2 + (v/4)*16 + half*8
            v16bf bv;
            const u16* bp = brow[t] + k0 + (half << 3);
#pragma unroll
            for (int v = 0; v < 8; ++v) {
                unsigned bw = *(const unsigned*)(bp + ((v & 3) << 1) + ((v >> 2) << 4));
                bv[2 * v]     = __builtin_bit_cast(__bf16, (u16)(bw & 0xFFFFu));
                bv[2 * v + 1] = __builtin_bit_cast(__bf16, (u16)(bw >> 16));
            }
            acc[t] = __builtin_amdgcn_wmma_f32_16x16x32_bf16(
                false, av, false, bv, (short)0, acc[t], false, false);
        }
    }

    // C layout: lane r = N, VGPR v -> M = v + half*8
#pragma unroll
    for (int t = 0; t < NT; ++t) {
        int cn  = tn * 64 + t * 16 + r;
        float bb = bias ? bias[cn] : 0.0f;
#pragma unroll
        for (int v = 0; v < 8; ++v) {
            int cm = tm * 16 + v + (half << 3);
            C[(size_t)cm * Nc + cn] = acc[t][v] + bb;
        }
    }
}

// per-(node,head) attention scores: as = <h[n,h,:], a_src[h,:]>, ad likewise
__global__ void k_attn_scores(const float* __restrict__ h,
                              const float* __restrict__ a_src, const float* __restrict__ a_dst,
                              float* __restrict__ as_, float* __restrict__ ad_,
                              int N, int H, int C) {
    int i = blockIdx.x * blockDim.x + threadIdx.x;
    if (i >= N * H) return;
    int n  = i / H;
    int hh = i - n * H;
    const float* hp = h + ((size_t)n * H + hh) * C;
    const float* s0 = a_src + hh * C;
    const float* d0 = a_dst + hh * C;
    float s = 0.f, d = 0.f;
    for (int c = 0; c < C; ++c) { float v = hp[c]; s += v * s0[c]; d += v * d0[c]; }
    as_[i] = s;
    ad_[i] = d;
}

// pass 1: per-dst running max of leaky_relu(as[src]+ad[dst]); edges then self-loops
__global__ void k_edge_max(const int* __restrict__ src, const int* __restrict__ dst,
                           int E, int N,
                           const float* __restrict__ as_, const float* __restrict__ ad_,
                           float* __restrict__ mx, int H) {
    long idx = (long)blockIdx.x * blockDim.x + threadIdx.x;
    long tot = (long)(E + N) * H;
    if (idx >= tot) return;
    int e  = (int)(idx / H);
    int hh = (int)(idx - (long)e * H);
    int s  = (e < E) ? src[e] : (e - E);
    int d  = (e < E) ? dst[e] : (e - E);
    float v = as_[s * H + hh] + ad_[d * H + hh];
    v = (v > 0.f) ? v : NEG_SLOPE * v;
    atomicMaxF32(&mx[d * H + hh], v);
}

// pass 2: ex = exp(v - mx[dst]); store per-edge, accumulate per-dst denom
__global__ void k_edge_expsum(const int* __restrict__ src, const int* __restrict__ dst,
                              int E, int N,
                              const float* __restrict__ as_, const float* __restrict__ ad_,
                              const float* __restrict__ mx,
                              float* __restrict__ exbuf, float* __restrict__ den, int H) {
    long idx = (long)blockIdx.x * blockDim.x + threadIdx.x;
    long tot = (long)(E + N) * H;
    if (idx >= tot) return;
    int e  = (int)(idx / H);
    int hh = (int)(idx - (long)e * H);
    int s  = (e < E) ? src[e] : (e - E);
    int d  = (e < E) ? dst[e] : (e - E);
    float v = as_[s * H + hh] + ad_[d * H + hh];
    v = (v > 0.f) ? v : NEG_SLOPE * v;
    float ex = expf(v - mx[d * H + hh]);
    exbuf[idx] = ex;
    atomicAdd(&den[d * H + hh], ex);
}

// pass 3: agg[dst,f] += (ex/den[dst]) * h[src,f]   (f = h*C + c)
__global__ void k_edge_agg(const int* __restrict__ src, const int* __restrict__ dst,
                           int E, int N,
                           const float* __restrict__ h,
                           const float* __restrict__ exbuf, const float* __restrict__ den,
                           float* __restrict__ agg, int H, int C) {
    int e = blockIdx.x;
    if (e >= E + N) return;
    int s = (e < E) ? src[e] : (e - E);
    int d = (e < E) ? dst[e] : (e - E);
    int F = H * C;
    const float* hs = h + (size_t)s * F;
    float*       ag = agg + (size_t)d * F;
    for (int f = threadIdx.x; f < F; f += blockDim.x) {
        int hh = f / C;
        float alpha = exbuf[(long)e * H + hh] / den[d * H + hh];
        atomicAdd(&ag[f], alpha * hs[f]);
    }
}

// epilogue: v = agg + bias, optional ELU; write optional f32 and/or bf16 outputs
__global__ void k_epilogue(const float* __restrict__ agg, const float* __restrict__ bias,
                           float* __restrict__ out32, u16* __restrict__ out16,
                           int N, int F, int do_elu) {
    long i = (long)blockIdx.x * blockDim.x + threadIdx.x;
    if (i >= (long)N * F) return;
    int f = (int)(i % F);
    float v = agg[i] + bias[f];
    if (do_elu) v = (v > 0.f) ? v : expm1f(v);
    if (out32) out32[i] = v;
    if (out16) out16[i] = f2bf_bits(v);
}

// ---------------------------------------------------------------- host side

static inline int cdivl(long a, long b) { return (int)((a + b - 1) / b); }

struct Scratch {
    u16*   featb16;  // [N,1024] bf16 layer input
    u16*   wb16;     // [1024,1024] bf16 W^T
    float* h32;      // [N,1024] projected features
    float* agg;      // [N,1024] aggregation
    float* as_;      // [N,8]
    float* ad_;      // [N,8]
    float* mx;       // [N,8]
    float* den;      // [N,8]
    float* exbuf;    // [(Emax+N),8]
};

static void run_gat(hipStream_t stream, const Scratch& S,
                    int N, int K, int H, int C,
                    const float* W, const float* a_src, const float* a_dst, const float* bias,
                    const int* ei, int E,
                    float* out32, u16* out16, int do_elu) {
    const int F = H * C;
    // 1) weights -> bf16, transposed to [F,K]
    {
        long n = (long)K * F;
        k_cast_transpose_bf16<<<cdivl(n, 256), 256, 0, stream>>>(W, S.wb16, K, F);
    }
    // 2) projection via WMMA: h = x_bf16 @ W_bf16  (16x64 tile per wave)
    {
        long waves = ((long)N / 16) * (F / 64);
        k_gemm_bf16_wmma<<<cdivl(waves * 32, 256), 256, 0, stream>>>(
            S.featb16, S.wb16, nullptr, S.h32, N, K, F);
    }
    // 3) attention scores
    k_attn_scores<<<cdivl((long)N * H, 256), 256, 0, stream>>>(
        S.h32, a_src, a_dst, S.as_, S.ad_, N, H, C);
    // 4) init reductions
    k_fill_f32<<<cdivl((long)N * H, 256), 256, 0, stream>>>(S.mx, -3.0e38f, (long)N * H);
    k_fill_f32<<<cdivl((long)N * H, 256), 256, 0, stream>>>(S.den, 0.0f, (long)N * H);
    k_fill_f32<<<cdivl((long)N * F, 256), 256, 0, stream>>>(S.agg, 0.0f, (long)N * F);
    // 5) softmax over incoming edges (+ self loops)
    const int* srcp = ei;
    const int* dstp = ei + E;
    long eh = (long)(E + N) * H;
    k_edge_max<<<cdivl(eh, 256), 256, 0, stream>>>(srcp, dstp, E, N, S.as_, S.ad_, S.mx, H);
    k_edge_expsum<<<cdivl(eh, 256), 256, 0, stream>>>(srcp, dstp, E, N, S.as_, S.ad_, S.mx,
                                                      S.exbuf, S.den, H);
    // 6) weighted aggregation
    int aggThreads = (F < 256) ? F : 256;
    k_edge_agg<<<E + N, aggThreads, 0, stream>>>(srcp, dstp, E, N, S.h32, S.exbuf, S.den,
                                                 S.agg, H, C);
    // 7) bias (+ELU), emit f32 and/or bf16
    k_epilogue<<<cdivl((long)N * F, 256), 256, 0, stream>>>(S.agg, bias, out32, out16,
                                                            N, F, do_elu);
}

extern "C" void kernel_launch(void* const* d_in, const int* in_sizes, int n_in,
                              void* d_out, int out_size, void* d_ws, size_t ws_size,
                              hipStream_t stream) {
    (void)n_in; (void)out_size; (void)ws_size;
    const int N = in_sizes[0] / 1024;   // 20000

    // ---- input indices (setup_inputs dict order, params depth-first W,a_src,a_dst,b)
    const float* x_gene  = (const float*)d_in[0];
    const float* x_mirna = (const float*)d_in[1];
    const float* x_to    = (const float*)d_in[2];
    const int GC1 = 3, GC2 = 7, GP = 11, GM = 15, GT = 19;
    const int MC1 = 23, MC2 = 27, MG = 31, MT = 35;
    const int TC1 = 39, TC2 = 43, TG = 47, TM = 51;
    const int FCW = 55, FCB = 56;
    const int EI_GG = 57, EI_GP = 58, EI_GM = 59, EI_GT = 60,
              EI_MM = 61, EI_MT = 62, EI_TT = 63;

    auto P  = [&](int i) { return (const float*)d_in[i]; };
    auto EI = [&](int i) { return (const int*)d_in[i]; };
    auto EC = [&](int i) { return in_sizes[i] / 2; };

    int Emax = 0;
    for (int i = EI_GG; i <= EI_TT; ++i) { int e = EC(i); if (e > Emax) Emax = e; }

    // ---- carve workspace
    char*  w   = (char*)d_ws;
    size_t off = 0;
    auto carve = [&](size_t bytes) { void* p = w + off; off = (off + bytes + 255) & ~(size_t)255; return p; };
    Scratch S;
    S.featb16 = (u16*)  carve((size_t)N * 1024 * sizeof(u16));
    S.wb16    = (u16*)  carve((size_t)1024 * 1024 * sizeof(u16));
    S.h32     = (float*)carve((size_t)N * 1024 * sizeof(float));
    S.agg     = (float*)carve((size_t)N * 1024 * sizeof(float));
    S.as_     = (float*)carve((size_t)N * 8 * sizeof(float));
    S.ad_     = (float*)carve((size_t)N * 8 * sizeof(float));
    S.mx      = (float*)carve((size_t)N * 8 * sizeof(float));
    S.den     = (float*)carve((size_t)N * 8 * sizeof(float));
    S.exbuf   = (float*)carve((size_t)(Emax + N) * 8 * sizeof(float));

    float* out_g = (float*)d_out;
    float* out_m = out_g + (size_t)N * 128;
    float* out_t = out_m + (size_t)N * 128;

    // ================= gene branch =================
    k_cast_bf16<<<cdivl((long)N * 1024, 256), 256, 0, stream>>>(x_gene, S.featb16, (long)N * 1024);
    run_gat(stream, S, N, 1024, 4, 256, P(GC1), P(GC1+1), P(GC1+2), P(GC1+3),
            EI(EI_GG), EC(EI_GG), nullptr, S.featb16, /*elu=*/1);
    run_gat(stream, S, N, 1024, 1, 256, P(GC2), P(GC2+1), P(GC2+2), P(GC2+3),
            EI(EI_GG), EC(EI_GG), nullptr, S.featb16, /*elu=*/0);
    run_gat(stream, S, N,  256, 1, 256, P(GP), P(GP+1), P(GP+2), P(GP+3),
            EI(EI_GP), EC(EI_GP), nullptr, S.featb16, /*elu=*/1);
    run_gat(stream, S, N,  256, 1, 256, P(GM), P(GM+1), P(GM+2), P(GM+3),
            EI(EI_GM), EC(EI_GM), nullptr, S.featb16, /*elu=*/1);
    run_gat(stream, S, N,  256, 1, 256, P(GT), P(GT+1), P(GT+2), P(GT+3),
            EI(EI_GT), EC(EI_GT), nullptr, S.featb16, /*elu=*/1);
    // fc: g = g @ W_fc + b_fc  -> out_g  (WMMA GEMM with fused bias)
    {
        long n = 256L * 128;
        k_cast_transpose_bf16<<<cdivl(n, 256), 256, 0, stream>>>(P(FCW), S.wb16, 256, 128);
        long waves = ((long)N / 16) * (128 / 64);
        k_gemm_bf16_wmma<<<cdivl(waves * 32, 256), 256, 0, stream>>>(
            S.featb16, S.wb16, P(FCB), out_g, N, 256, 128);
    }

    // ================= miRNA branch =================
    k_cast_bf16<<<cdivl((long)N * 512, 256), 256, 0, stream>>>(x_mirna, S.featb16, (long)N * 512);
    run_gat(stream, S, N,  512, 8, 128, P(MC1), P(MC1+1), P(MC1+2), P(MC1+3),
            EI(EI_MM), EC(EI_MM), nullptr, S.featb16, /*elu=*/1);
    run_gat(stream, S, N, 1024, 1, 128, P(MC2), P(MC2+1), P(MC2+2), P(MC2+3),
            EI(EI_MM), EC(EI_MM), nullptr, S.featb16, /*elu=*/0);
    run_gat(stream, S, N,  128, 1, 128, P(MG), P(MG+1), P(MG+2), P(MG+3),
            EI(EI_GM), EC(EI_GM), nullptr, S.featb16, /*elu=*/1);
    run_gat(stream, S, N,  128, 1, 128, P(MT), P(MT+1), P(MT+2), P(MT+3),
            EI(EI_MT), EC(EI_MT), out_m, nullptr, /*elu=*/0);

    // ================= TO branch =================
    k_cast_bf16<<<cdivl((long)N * 512, 256), 256, 0, stream>>>(x_to, S.featb16, (long)N * 512);
    run_gat(stream, S, N,  512, 4, 256, P(TC1), P(TC1+1), P(TC1+2), P(TC1+3),
            EI(EI_TT), EC(EI_TT), nullptr, S.featb16, /*elu=*/1);
    run_gat(stream, S, N, 1024, 1, 128, P(TC2), P(TC2+1), P(TC2+2), P(TC2+3),
            EI(EI_TT), EC(EI_TT), nullptr, S.featb16, /*elu=*/0);
    run_gat(stream, S, N,  128, 1, 128, P(TG), P(TG+1), P(TG+2), P(TG+3),
            EI(EI_GT), EC(EI_GT), nullptr, S.featb16, /*elu=*/1);
    run_gat(stream, S, N,  128, 1, 128, P(TM), P(TM+1), P(TM+2), P(TM+3),
            EI(EI_MT), EC(EI_MT), out_t, nullptr, /*elu=*/1);
}
